// Int8DynActInt8WeightLinear_55783035241143
// MI455X (gfx1250) — compile-verified, matrix operand order checked by reading, not run on hardware
//
#include <hip/hip_runtime.h>
#include <stdint.h>

typedef __attribute__((ext_vector_type(8))) int v8i;

// ---------------------------------------------------------------------------
// Kernel 1: per-token dynamic asymmetric int8 quantization (one wave32/token)
//   writes: q (int8, row-major [M][K]), sx[m]=scale, zp[m]=zero-point (float),
//           sumq[m] = sum_i q[m,i]
// ---------------------------------------------------------------------------
__global__ __launch_bounds__(256) void quant_tokens_kernel(
    const float* __restrict__ x, int8_t* __restrict__ q,
    float* __restrict__ sx, float* __restrict__ zp, int* __restrict__ sumq,
    int K) {
  const int lane = threadIdx.x & 31;
  const int wave = threadIdx.x >> 5;
  const int m = blockIdx.x * 8 + wave;

  const float* xr = x + (size_t)m * K;
  const int nv4 = K >> 7;  // float4 chunks per lane (K/32/4)

  float mn = 0.0f, mx = 0.0f;  // reference clamps range to include 0
#pragma unroll 4
  for (int it = 0; it < nv4; ++it) {
    float4 v = ((const float4*)xr)[it * 32 + lane];
    mn = fminf(mn, fminf(fminf(v.x, v.y), fminf(v.z, v.w)));
    mx = fmaxf(mx, fmaxf(fmaxf(v.x, v.y), fmaxf(v.z, v.w)));
  }
#pragma unroll
  for (int off = 16; off > 0; off >>= 1) {
    mn = fminf(mn, __shfl_xor(mn, off, 32));
    mx = fmaxf(mx, __shfl_xor(mx, off, 32));
  }

  const float eps = 1.1920928955078125e-07f;  // FLT_EPSILON == finfo(f32).eps
  float scale = fmaxf((mx - mn) * (1.0f / 255.0f), eps);
  float zpf = -128.0f - rintf(mn / scale);            // QMIN - round(mn/scale)
  zpf = fminf(fmaxf(zpf, -128.0f), 127.0f);

  int8_t* qr = q + (size_t)m * K;
  int sq = 0;
#pragma unroll 4
  for (int it = 0; it < nv4; ++it) {
    float4 v = ((const float4*)xr)[it * 32 + lane];
    // true division to match jnp.round(x / scale) RNE ties exactly
    int q0 = (int)fminf(fmaxf(rintf(v.x / scale) + zpf, -128.0f), 127.0f);
    int q1 = (int)fminf(fmaxf(rintf(v.y / scale) + zpf, -128.0f), 127.0f);
    int q2 = (int)fminf(fmaxf(rintf(v.z / scale) + zpf, -128.0f), 127.0f);
    int q3 = (int)fminf(fmaxf(rintf(v.w / scale) + zpf, -128.0f), 127.0f);
    sq += q0 + q1 + q2 + q3;
    uint32_t pk = (uint32_t)(q0 & 0xff) | ((uint32_t)(q1 & 0xff) << 8) |
                  ((uint32_t)(q2 & 0xff) << 16) | ((uint32_t)(q3 & 0xff) << 24);
    ((uint32_t*)qr)[it * 32 + lane] = pk;
  }
#pragma unroll
  for (int off = 16; off > 0; off >>= 1) sq += __shfl_xor(sq, off, 32);

  if (lane == 0) {
    sx[m] = scale;
    zp[m] = zpf;
    sumq[m] = sq;
  }
}

// ---------------------------------------------------------------------------
// Kernel 2: per-output-channel weight row sums rw[o] = sum_i w[o,i]
// ---------------------------------------------------------------------------
__device__ __forceinline__ int sum4b(int v) {
  return (int)(int8_t)v + (int)(int8_t)(v >> 8) + (int)(int8_t)(v >> 16) +
         (v >> 24);
}

__global__ __launch_bounds__(256) void weight_rowsum_kernel(
    const int8_t* __restrict__ w, int* __restrict__ rw, int K) {
  const int lane = threadIdx.x & 31;
  const int wave = threadIdx.x >> 5;
  const int o = blockIdx.x * 8 + wave;

  const int4* wr = (const int4*)(w + (size_t)o * K);
  const int nv = K >> 9;  // int4 (16B) chunks per lane
  int s = 0;
#pragma unroll
  for (int it = 0; it < nv; ++it) {
    int4 v = wr[it * 32 + lane];
    s += sum4b(v.x) + sum4b(v.y) + sum4b(v.z) + sum4b(v.w);
  }
#pragma unroll
  for (int off = 16; off > 0; off >>= 1) s += __shfl_xor(s, off, 32);
  if (lane == 0) rw[o] = s;
}

// ---------------------------------------------------------------------------
// Kernel 3: int8 WMMA GEMM + dequant epilogue.
//   Block tile 256(M) x 128(N); 8 waves; each wave 64x64 = 4x4 16x16 tiles.
//   (48 B of tile traffic per output element -> ~1.6 GB total from L2,
//    2 global_load_b64 per v_wmma_i32_16x16x64_iu8.)
//   A/B fragments loaded directly in the CDNA5 8-bit WMMA lane layout:
//     lanes 0-15: row = lane&15, K-chunks at k0 + {0,16,32,48}
//     lanes 16-31: same rows, K-chunks at k0 + 8 + {0,16,32,48}
//   (each chunk = 8 bytes -> 2 dwords of the v8i fragment)
// ---------------------------------------------------------------------------
__device__ __forceinline__ v8i load_frag8(const int8_t* p) {
  int2 c0 = *(const int2*)(p);
  int2 c1 = *(const int2*)(p + 16);
  int2 c2 = *(const int2*)(p + 32);
  int2 c3 = *(const int2*)(p + 48);
  v8i f;
  f[0] = c0.x; f[1] = c0.y;
  f[2] = c1.x; f[3] = c1.y;
  f[4] = c2.x; f[5] = c2.y;
  f[6] = c3.x; f[7] = c3.y;
  return f;
}

__global__ __launch_bounds__(256) void int8_wmma_gemm_kernel(
    const int8_t* __restrict__ q, const int8_t* __restrict__ w,
    const float* __restrict__ sx, const float* __restrict__ zp,
    const int* __restrict__ sumq, const int* __restrict__ rw,
    const float* __restrict__ scales, const float* __restrict__ zeros,
    float* __restrict__ out, int M, int N, int K) {
  const int lane = threadIdx.x & 31;
  const int wave = threadIdx.x >> 5;
  const int wm = wave & 3;   // 4 waves tile M (4 x 64 = 256)
  const int wn = wave >> 2;  // 2 waves tile N (2 x 64 = 128)

  const int mBase = blockIdx.y * 256 + wm * 64;
  const int nBase = blockIdx.x * 128 + wn * 64;

  const int lr = lane & 15;
  const int khalf = (lane >> 4) << 3;  // +8 bytes for upper half-wave

  const int8_t* aPtr[4];
#pragma unroll
  for (int mt = 0; mt < 4; ++mt)
    aPtr[mt] = q + (size_t)(mBase + mt * 16 + lr) * K + khalf;
  const int8_t* bPtr[4];
#pragma unroll
  for (int nt = 0; nt < 4; ++nt)
    bPtr[nt] = w + (size_t)(nBase + nt * 16 + lr) * K + khalf;

  v8i acc[4][4];
#pragma unroll
  for (int mt = 0; mt < 4; ++mt)
#pragma unroll
    for (int nt = 0; nt < 4; ++nt) acc[mt][nt] = v8i{0, 0, 0, 0, 0, 0, 0, 0};

  for (int k0 = 0; k0 < K; k0 += 64) {
    v8i a[4], b[4];
#pragma unroll
    for (int mt = 0; mt < 4; ++mt) a[mt] = load_frag8(aPtr[mt] + k0);
#pragma unroll
    for (int nt = 0; nt < 4; ++nt) b[nt] = load_frag8(bPtr[nt] + k0);
#pragma unroll
    for (int mt = 0; mt < 4; ++mt)
#pragma unroll
      for (int nt = 0; nt < 4; ++nt)
        acc[mt][nt] = __builtin_amdgcn_wmma_i32_16x16x64_iu8(
            /*sgn_a=*/true, a[mt], /*sgn_b=*/true, b[nt], acc[mt][nt],
            /*reuse_a=*/false, /*reuse_b=*/false);
  }

  // Epilogue: out[m,n] = sx[m]*scales[n]*(acc - zp[m]*rw[n]
  //                                       - zeros[n]*(sumq[m] - K*zp[m]))
  const int hi8 = (lane >> 4) * 8;
  const float Kf = (float)K;
#pragma unroll
  for (int mt = 0; mt < 4; ++mt) {
    float sxm[8], zpm[8], sqm[8];
#pragma unroll
    for (int r = 0; r < 8; ++r) {
      int m = mBase + mt * 16 + r + hi8;
      sxm[r] = sx[m];
      zpm[r] = zp[m];
      sqm[r] = (float)sumq[m];
    }
#pragma unroll
    for (int nt = 0; nt < 4; ++nt) {
      int n = nBase + nt * 16 + lr;
      float scn = scales[n];
      float zrn = zeros[n];
      float rwn = (float)rw[n];
#pragma unroll
      for (int r = 0; r < 8; ++r) {
        int m = mBase + mt * 16 + r + hi8;
        float t = (float)acc[mt][nt][r] - zpm[r] * rwn -
                  zrn * (sqm[r] - Kf * zpm[r]);
        out[(size_t)m * N + n] = sxm[r] * scn * t;
      }
    }
  }
}

// ---------------------------------------------------------------------------
// Launch
// ---------------------------------------------------------------------------
extern "C" void kernel_launch(void* const* d_in, const int* in_sizes, int n_in,
                              void* d_out, int out_size, void* d_ws,
                              size_t ws_size, hipStream_t stream) {
  (void)n_in; (void)out_size; (void)ws_size;

  const float* x = (const float*)d_in[0];
  const int8_t* w = (const int8_t*)d_in[1];
  const float* scales = (const float*)d_in[2];
  const float* zeros = (const float*)d_in[3];
  float* out = (float*)d_out;

  const int OUT = in_sizes[2];           // 4096
  const int IN = in_sizes[1] / OUT;      // 4096
  const int M = in_sizes[0] / IN;        // B*S = 8192

  // workspace layout
  int8_t* q = (int8_t*)d_ws;                       // M*IN int8
  float* sx = (float*)(q + (size_t)M * IN);        // M
  float* zp = sx + M;                              // M
  int* sumq = (int*)(zp + M);                      // M
  int* rw = sumq + M;                              // OUT

  quant_tokens_kernel<<<M / 8, 256, 0, stream>>>(x, q, sx, zp, sumq, IN);
  weight_rowsum_kernel<<<OUT / 8, 256, 0, stream>>>(w, rw, IN);

  dim3 grid(OUT / 128, M / 256);
  int8_wmma_gemm_kernel<<<grid, 256, 0, stream>>>(q, w, sx, zp, sumq, rw,
                                                  scales, zeros, out, M, OUT,
                                                  IN);
}
